// LSTransformerHead_18683107737676
// MI455X (gfx1250) — matrix operand order repaired and checked by reading, not compiled, not run on hardware
//
#include <hip/hip_runtime.h>
#include <math.h>

// ---------------------------------------------------------------------------
// Types matching gfx1250 WMMA builtin signatures
// ---------------------------------------------------------------------------
typedef __bf16 bf16;
typedef __bf16 v16bf __attribute__((ext_vector_type(16)));
typedef __bf16 v8bf  __attribute__((ext_vector_type(8)));
typedef float  v8f   __attribute__((ext_vector_type(8)));

// ---------------------------------------------------------------------------
// Problem constants (from the reference)
// ---------------------------------------------------------------------------
constexpr int B_  = 4, N_ = 4096, V_ = 512, D_ = 512, H_ = 8, DH_ = 64;
constexpr int L_  = 4, WS_ = 128, R_ = 256, C_ = 10;
constexpr int FF_ = 4 * D_;              // 2048
constexpr int NW_ = N_ / WS_;            // 32 windows
constexpr int HD_ = H_ * DH_;            // 512
constexpr long long NV_ = (long long)N_ * V_;   // 2097152
constexpr float SCALE_ = 0.125f;         // DH^-0.5

// GEMM epilogue / mode flags
enum { F_BIAS = 1, F_RES = 2, F_GELU = 4, F_BF16OUT = 8, F_BNN = 16 };

struct GemmArgs {
  const bf16* A;        // (M x K) row-major, bf16.  M % 128 == 0 for every launch.
  const bf16* Bm;       // NT: (Nn x K) row-major (Nn%128==0); NN: (K x Nn) row-major
  float*      Cf;       // f32 out
  bf16*       Cb;       // bf16 out
  const float* bias;    // per-column
  const float* resid;   // residual, same layout as C (f32)
  int M, Nn, K;
  int lda, ldb, ldc;
  int flags;
  float alpha;
  // batch: z -> z0 = z%bn0, z1 = (z/bn0)%bn1, z2 = z/(bn0*bn1)
  int bn0, bn1;
  long long sA0, sA1, sA2;
  long long sB0, sB1, sB2;
  long long sC0, sC1, sC2;
};

__device__ __forceinline__ float gelu_tanh(float x) {
  const float k0 = 0.7978845608028654f;   // sqrt(2/pi)
  const float k1 = 0.044715f;
  float u = k0 * (x + k1 * x * x * x);
  return 0.5f * x * (1.0f + tanhf(u));
}

__device__ __forceinline__ v16bf pack16(const bf16* lo, const bf16* hi) {
  v8bf a = *reinterpret_cast<const v8bf*>(lo);
  v8bf b = *reinterpret_cast<const v8bf*>(hi);
  v16bf r;
#pragma unroll
  for (int i = 0; i < 8; ++i) { r[i] = a[i]; r[i + 8] = b[i]; }
  return r;
}

// ---------------------------------------------------------------------------
// gfx1250 async global->LDS copy (ISA 10.7 / 15.18: GLOBAL_LOAD_ASYNC_TO_LDS_B128,
// tracked with ASYNCcnt).  VDST VGPR carries the per-lane LDS byte address,
// VADDR the 64-bit flat address.  Completion enforced with s_wait_asynccnt 0.
// ---------------------------------------------------------------------------
__device__ __forceinline__ unsigned to_lds_off(const void* p) {
  // generic -> addrspace(3) cast yields the 32-bit LDS byte offset
  return (unsigned)(unsigned long long)
         (__attribute__((address_space(3))) const void*)p;
}
__device__ __forceinline__ void async_copy_b128(unsigned lds_off, const void* gptr) {
  asm volatile("global_load_async_to_lds_b128 %0, %1, off"
               :: "v"(lds_off), "v"((unsigned long long)(uintptr_t)gptr)
               : "memory");
}
__device__ __forceinline__ void wait_async0() {
  asm volatile("s_wait_asynccnt 0x0" ::: "memory");
}

// ---------------------------------------------------------------------------
// Batched bf16 WMMA GEMM.  Block tile 128x128, BK=32 (= WMMA K-depth).
// 8 wave32 waves, each owns a 32x64 sub-tile = 2x4 WMMA accumulators.
// Fragment layouts per CDNA5 ISA 7.12.2:
//   A (16x32 bf16): lane = 16*Khalf + M ; elems[0..7]=K[Khalf*8..], [8..15]=K[16+Khalf*8..]
//   B (32x16 bf16): lane = 16*Khalf + N ; elems[0..7]=K[Khalf*8..], [8..15]=K[16+Khalf*8..]
//   C/D (16x16 f32): vgpr e -> M = e + 8*(lane>=16), N = lane%16
// ---------------------------------------------------------------------------
__global__ __launch_bounds__(256) void gemm_bf16_kernel(GemmArgs g) {
  constexpr int BM = 128, BN = 128, BK = 32, PAD = 8;
  __shared__ __attribute__((aligned(16))) bf16 As[BM][BK + PAD];
  __shared__ __attribute__((aligned(16))) bf16 Bs[BN][BK + PAD];

  const int z  = blockIdx.z;
  const int z0 = z % g.bn0;
  const int zq = z / g.bn0;
  const int z1 = zq % g.bn1;
  const int z2 = zq / g.bn1;

  const bf16* A  = g.A  + (long long)z0 * g.sA0 + (long long)z1 * g.sA1 + (long long)z2 * g.sA2;
  const bf16* Bm = g.Bm + (long long)z0 * g.sB0 + (long long)z1 * g.sB1 + (long long)z2 * g.sB2;
  const long long cbase = (long long)z0 * g.sC0 + (long long)z1 * g.sC1 + (long long)z2 * g.sC2;

  const int m0 = blockIdx.y * BM;
  const int n0 = blockIdx.x * BN;
  const int tid  = threadIdx.x;
  const int wave = tid >> 5, lane = tid & 31;
  const int wr = wave >> 1, wc = wave & 1;       // 4x2 wave grid
  const int lm = lane & 15, lh = lane >> 4;

  v8f acc[2][4];
#pragma unroll
  for (int i = 0; i < 2; ++i)
#pragma unroll
    for (int j = 0; j < 4; ++j)
#pragma unroll
      for (int e = 0; e < 8; ++e) acc[i][j][e] = 0.0f;

  // Per-thread staging coordinates: 2 threads per tile row, 16 bf16 (2x b128) each.
  const int srow = tid >> 1;
  const int sseg = (tid & 1) << 4;

  for (int k0 = 0; k0 < g.K; k0 += BK) {
    { // A tile (BM x BK): always fully in-bounds (M % 128 == 0) -> async copy
      const bf16* src = A + (long long)(m0 + srow) * g.lda + k0 + sseg;
      async_copy_b128(to_lds_off(&As[srow][sseg]),     src);
      async_copy_b128(to_lds_off(&As[srow][sseg + 8]), src + 8);
      if (k0 + BK < g.K) __builtin_prefetch(src + BK, 0, 3);  // global_prefetch_b8
    }
    if (!(g.flags & F_BNN)) {
      // B is (Nn x K) with Nn % 128 == 0: async copy, no guard needed
      const bf16* src = Bm + (long long)(n0 + srow) * g.ldb + k0 + sseg;
      async_copy_b128(to_lds_off(&Bs[srow][sseg]),     src);
      async_copy_b128(to_lds_off(&Bs[srow][sseg + 8]), src + 8);
      if (k0 + BK < g.K) __builtin_prefetch(src + BK, 0, 3);
    } else {
      // B is (K x Nn): transpose while staging. 16-column group is fully
      // in- or out-of-bounds whenever Nn % 16 == 0 (true for all launches).
      const int kk = tid & 31;
      const int nb = (tid >> 5) << 4;
      const bf16* src = Bm + (long long)(k0 + kk) * g.ldb + n0 + nb;
      bf16 tmp[16];
      if (n0 + nb + 16 <= g.Nn) {
        *reinterpret_cast<uint4*>(&tmp[0]) = *reinterpret_cast<const uint4*>(src);
        *reinterpret_cast<uint4*>(&tmp[8]) = *reinterpret_cast<const uint4*>(src + 8);
      } else {
#pragma unroll
        for (int i = 0; i < 16; ++i)
          tmp[i] = (n0 + nb + i < g.Nn) ? src[i] : (bf16)0.0f;
      }
#pragma unroll
      for (int i = 0; i < 16; ++i) Bs[nb + i][kk] = tmp[i];
    }
    wait_async0();        // ASYNCcnt == 0: all our async LDS writes have landed
    __syncthreads();      // (also waits DScnt for the NN ds_store path)

    v16bf af[2];
#pragma unroll
    for (int i = 0; i < 2; ++i) {
      const int row = wr * 32 + i * 16 + lm;
      af[i] = pack16(&As[row][lh * 8], &As[row][16 + lh * 8]);
    }
    v16bf bfv[4];
#pragma unroll
    for (int j = 0; j < 4; ++j) {
      const int col = wc * 64 + j * 16 + lm;
      bfv[j] = pack16(&Bs[col][lh * 8], &Bs[col][16 + lh * 8]);
    }
#pragma unroll
    for (int i = 0; i < 2; ++i)
#pragma unroll
      for (int j = 0; j < 4; ++j)
        acc[i][j] = __builtin_amdgcn_wmma_f32_16x16x32_bf16(
            false, af[i], false, bfv[j], (short)0, acc[i][j], false, false);
    __syncthreads();
  }

  // Epilogue: alpha, bias, gelu, residual, bf16/f32 store.
  // mrow is always < M (M % 128 == 0); only ncol needs a guard (NN: Nn=64).
#pragma unroll
  for (int j = 0; j < 4; ++j) {
    const int ncol = n0 + wc * 64 + j * 16 + lm;
    if (ncol >= g.Nn) continue;
#pragma unroll
    for (int i = 0; i < 2; ++i) {
#pragma unroll
      for (int e = 0; e < 8; ++e) {
        const int mrow = m0 + wr * 32 + i * 16 + lh * 8 + e;
        float v = acc[i][j][e] * g.alpha;
        if (g.flags & F_BIAS) v += g.bias[ncol];
        if (g.flags & F_GELU) v = gelu_tanh(v);
        const long long off = cbase + (long long)mrow * g.ldc + ncol;
        if (g.flags & F_RES) v += g.resid[off];
        if (g.flags & F_BF16OUT) g.Cb[off] = (bf16)v;
        else                     g.Cf[off] = v;
      }
    }
  }
}

// ---------------------------------------------------------------------------
// Elementwise / reduction kernels (f32 VALU)
// ---------------------------------------------------------------------------
__global__ void embed_kernel(const int* x, const float* tok, const float* pos, float* h) {
  const int row = blockIdx.x;          // b*N + n
  const int n = row % N_;
  const long long t = x[row];
  const float* te = tok + t * D_;
  const float* pe = pos + (long long)n * D_;
  float* hp = h + (long long)row * D_;
  for (int c = threadIdx.x; c < D_; c += blockDim.x) hp[c] = te[c] + pe[c];
}

// offset(row) = row*inStride + (row/rowsPerGroup)*groupSkip   (handles k/v interleave)
__global__ void layernorm_kernel(const float* in, bf16* out, const float* gamma,
                                 const float* beta, int cols, int rowsPerGroup,
                                 long long groupSkip, int inStride, int outStride) {
  __shared__ float red[256];
  const int row = blockIdx.x;
  const long long ioff = (long long)row * inStride + (long long)(row / rowsPerGroup) * groupSkip;
  const float* xp = in + ioff;
  const int t = threadIdx.x, bd = blockDim.x;
  float s = 0.f, ss = 0.f;
  for (int c = t; c < cols; c += bd) { float v = xp[c]; s += v; ss += v * v; }
  red[t] = s; __syncthreads();
  for (int o = bd >> 1; o > 0; o >>= 1) { if (t < o) red[t] += red[t + o]; __syncthreads(); }
  const float mean = red[0] / cols; __syncthreads();
  red[t] = ss; __syncthreads();
  for (int o = bd >> 1; o > 0; o >>= 1) { if (t < o) red[t] += red[t + o]; __syncthreads(); }
  const float var = red[0] / cols - mean * mean;
  const float inv = rsqrtf(var + 1e-5f);
  bf16* yp = out + (long long)row * outStride;
  for (int c = t; c < cols; c += bd)
    yp[c] = (bf16)((xp[c] - mean) * inv * gamma[c] + beta[c]);
}

__global__ void softmax_rows_kernel(const float* in, bf16* out, int cols) {
  __shared__ float red[256];
  const long long base = (long long)blockIdx.x * cols;
  const int t = threadIdx.x, bd = blockDim.x;
  float lv[16];
  int cnt = 0;
  float mx = -INFINITY;
  for (int c = t; c < cols; c += bd) { float v = in[base + c]; lv[cnt++] = v; mx = fmaxf(mx, v); }
  red[t] = mx; __syncthreads();
  for (int o = bd >> 1; o > 0; o >>= 1) { if (t < o) red[t] = fmaxf(red[t], red[t + o]); __syncthreads(); }
  mx = red[0]; __syncthreads();
  float s = 0.f;
  for (int i = 0; i < cnt; ++i) { lv[i] = __expf(lv[i] - mx); s += lv[i]; }
  red[t] = s; __syncthreads();
  for (int o = bd >> 1; o > 0; o >>= 1) { if (t < o) red[t] += red[t + o]; __syncthreads(); }
  const float inv = 1.0f / red[0];
  cnt = 0;
  for (int c = t; c < cols; c += bd) out[base + c] = (bf16)(lv[cnt++] * inv);
}

__global__ void f32_to_bf16_kernel(const float* in, bf16* out, long long n) {
  for (long long i = (long long)blockIdx.x * blockDim.x + threadIdx.x; i < n;
       i += (long long)gridDim.x * blockDim.x)
    out[i] = (bf16)in[i];
}

// out[n*K + k] = in[k*Ncols + n]  (weights -> bf16, transposed to (N x K))
__global__ void transpose_f32_to_bf16_kernel(const float* in, bf16* out, int K, int Ncols) {
  const long long total = (long long)K * Ncols;
  for (long long i = (long long)blockIdx.x * blockDim.x + threadIdx.x; i < total;
       i += (long long)gridDim.x * blockDim.x) {
    const int k = (int)(i / Ncols);
    const int n = (int)(i % Ncols);
    out[(long long)n * K + k] = (bf16)in[i];
  }
}

// Final head: out[b,c] = sum_j logits[b,j]*Wfin[j,c] + bfin[c], deterministic 2-stage
constexpr int FCHUNKS = 256;
constexpr int FCHUNK_J = (int)(NV_ / FCHUNKS);   // 8192
__global__ void final_partial_kernel(const float* logits, const float* Wfin, float* part) {
  __shared__ float red[256 * 40];
  const int cid = blockIdx.x, t = threadIdx.x;
  float acc[40];
#pragma unroll
  for (int k = 0; k < 40; ++k) acc[k] = 0.f;
  const long long j0 = (long long)cid * FCHUNK_J;
  for (int jj = t; jj < FCHUNK_J; jj += 256) {
    const long long j = j0 + jj;
    const float* wrow = Wfin + j * C_;
    const float l0 = logits[j];
    const float l1 = logits[NV_ + j];
    const float l2 = logits[2 * NV_ + j];
    const float l3 = logits[3 * NV_ + j];
#pragma unroll
    for (int c = 0; c < C_; ++c) {
      const float w = wrow[c];
      acc[c]      += l0 * w;
      acc[10 + c] += l1 * w;
      acc[20 + c] += l2 * w;
      acc[30 + c] += l3 * w;
    }
  }
#pragma unroll
  for (int k = 0; k < 40; ++k) red[t * 40 + k] = acc[k];
  __syncthreads();
  for (int o = 128; o > 0; o >>= 1) {
    if (t < o)
#pragma unroll
      for (int k = 0; k < 40; ++k) red[t * 40 + k] += red[(t + o) * 40 + k];
    __syncthreads();
  }
  if (t < 40) part[cid * 40 + t] = red[t];
}

__global__ void final_reduce_kernel(const float* part, const float* bfin, float* out) {
  const int t = threadIdx.x;
  if (t < 40) {
    float s = bfin[t % C_];
    for (int ch = 0; ch < FCHUNKS; ++ch) s += part[ch * 40 + t];
    out[t] = s;
  }
}

// ---------------------------------------------------------------------------
// Host side
// ---------------------------------------------------------------------------
static inline GemmArgs ga_default() {
  GemmArgs g{};
  g.bn0 = 1; g.bn1 = 1; g.alpha = 1.0f;
  return g;
}
static inline void launch_gemm(hipStream_t stream, const GemmArgs& g, int batch) {
  dim3 grid((g.Nn + 127) / 128, (g.M + 127) / 128, batch);
  gemm_bf16_kernel<<<grid, dim3(256), 0, stream>>>(g);
}

extern "C" void kernel_launch(void* const* d_in, const int* in_sizes, int n_in,
                              void* d_out, int out_size, void* d_ws, size_t ws_size,
                              hipStream_t stream) {
  (void)in_sizes; (void)n_in; (void)out_size; (void)ws_size;

  const int*   x       = (const int*)  d_in[0];
  const float* tok_emb = (const float*)d_in[1];
  const float* pos_emb = (const float*)d_in[2];
  const float* ln1_g   = (const float*)d_in[3];
  const float* ln1_b   = (const float*)d_in[4];
  const float* Wq      = (const float*)d_in[5];
  const float* Wkv     = (const float*)d_in[6];
  const float* Wp      = (const float*)d_in[7];
  const float* lln_g   = (const float*)d_in[8];
  const float* lln_b   = (const float*)d_in[9];
  const float* gln_g   = (const float*)d_in[10];
  const float* gln_b   = (const float*)d_in[11];
  const float* Wo      = (const float*)d_in[12];
  const float* bo      = (const float*)d_in[13];
  const float* ln2_g   = (const float*)d_in[14];
  const float* ln2_b   = (const float*)d_in[15];
  const float* W1      = (const float*)d_in[16];
  const float* b1      = (const float*)d_in[17];
  const float* W2      = (const float*)d_in[18];
  const float* b2      = (const float*)d_in[19];
  const float* lnf_g   = (const float*)d_in[20];
  const float* lnf_b   = (const float*)d_in[21];
  const float* Wlog    = (const float*)d_in[22];
  const float* blog    = (const float*)d_in[23];
  const float* Wfin    = (const float*)d_in[24];
  const float* bfin    = (const float*)d_in[25];
  float* out = (float*)d_out;

  // ---- workspace bump allocator (total ~590 MB with aliasing) ----
  char* base = (char*)d_ws;
  size_t off = 0;
  auto alloc = [&](size_t bytes) -> void* {
    void* p = base + off;
    off += (bytes + 255) & ~(size_t)255;
    return p;
  };
  float* h    = (float*)alloc((size_t)B_ * N_ * D_ * 4);
  bf16*  y    = (bf16*) alloc((size_t)B_ * N_ * D_ * 2);
  bf16*  qb   = (bf16*) alloc((size_t)B_ * N_ * HD_ * 2);
  float* kv   = (float*)alloc((size_t)B_ * N_ * 2 * HD_ * 4);
  bf16*  kvb  = (bf16*) alloc((size_t)B_ * N_ * 2 * HD_ * 2);
  void*  P5   = alloc((size_t)B_ * H_ * N_ * (WS_ + R_) * 4); // sim f32 (largest)
  void*  P6   = alloc((size_t)B_ * H_ * N_ * (WS_ + R_) * 2); // attn bf16
  float* sT   = (float*)P5;   // (B,H,R,N) scores; dead before sim is written
  float* sim  = (float*)P5;
  bf16*  a1   = (bf16*)P5;    // FFN hidden; alive only after attention is consumed
  float* logits = (float*)P5; // alive only after the last layer
  bf16*  pT   = (bf16*)P6;    // dead before attn is written
  bf16*  attn = (bf16*)P6;
  float* gkraw = (float*)alloc((size_t)B_ * H_ * R_ * DH_ * 4);
  float* gvraw = (float*)alloc((size_t)B_ * H_ * R_ * DH_ * 4);
  bf16*  gkb   = (bf16*) alloc((size_t)B_ * H_ * R_ * DH_ * 2);
  bf16*  gvb   = (bf16*) alloc((size_t)B_ * H_ * R_ * DH_ * 2);
  bf16*  lk    = (bf16*) alloc((size_t)B_ * N_ * HD_ * 2);
  bf16*  lv    = (bf16*) alloc((size_t)B_ * N_ * HD_ * 2);
  float* o     = (float*)alloc((size_t)B_ * N_ * HD_ * 4);
  bf16*  ob    = (bf16*) alloc((size_t)B_ * N_ * HD_ * 2);
  float* part  = (float*)alloc((size_t)FCHUNKS * 40 * 4);
  bf16 *WqT[L_], *WkvT[L_], *WpT[L_], *WoT[L_], *W1T[L_], *W2T[L_];
  for (int l = 0; l < L_; ++l) {
    WqT[l]  = (bf16*)alloc((size_t)D_ * HD_ * 2);
    WkvT[l] = (bf16*)alloc((size_t)D_ * 2 * HD_ * 2);
    WpT[l]  = (bf16*)alloc((size_t)DH_ * R_ * 2);
    WoT[l]  = (bf16*)alloc((size_t)HD_ * D_ * 2);
    W1T[l]  = (bf16*)alloc((size_t)D_ * FF_ * 2);
    W2T[l]  = (bf16*)alloc((size_t)FF_ * D_ * 2);
  }
  bf16* WlogT = (bf16*)alloc((size_t)D_ * V_ * 2);

  // ---- weights -> bf16, transposed to (N x K) for the NT GEMM B operand ----
  auto tr = [&](const float* in, bf16* outp, int K, int Ncols) {
    transpose_f32_to_bf16_kernel<<<2048, 256, 0, stream>>>(in, outp, K, Ncols);
  };
  for (int l = 0; l < L_; ++l) {
    tr(Wq  + (size_t)l * D_ * HD_,     WqT[l],  D_,  HD_);
    tr(Wkv + (size_t)l * D_ * 2 * HD_, WkvT[l], D_,  2 * HD_);
    tr(Wp  + (size_t)l * DH_ * R_,     WpT[l],  DH_, R_);
    tr(Wo  + (size_t)l * HD_ * D_,     WoT[l],  HD_, D_);
    tr(W1  + (size_t)l * D_ * FF_,     W1T[l],  D_,  FF_);
    tr(W2  + (size_t)l * FF_ * D_,     W2T[l],  FF_, D_);
  }
  tr(Wlog, WlogT, D_, V_);

  // ---- embedding ----
  embed_kernel<<<B_ * N_, 256, 0, stream>>>(x, tok_emb, pos_emb, h);

  const int M = B_ * N_;          // 16384 token rows
  for (int l = 0; l < L_; ++l) {
    // y = LN1(h)  [bf16]
    layernorm_kernel<<<M, 256, 0, stream>>>(h, y, ln1_g + l * D_, ln1_b + l * D_,
                                            D_, 1, 0, D_, D_);
    // q = SCALE * y @ Wq   -> bf16 (B,N,H,DH)
    { GemmArgs g = ga_default(); g.A = y; g.Bm = WqT[l]; g.Cb = qb;
      g.M = M; g.Nn = HD_; g.K = D_; g.lda = D_; g.ldb = D_; g.ldc = HD_;
      g.flags = F_BF16OUT; g.alpha = SCALE_; launch_gemm(stream, g, 1); }
    // kv = y @ Wkv  -> f32 (B,N,2*H*DH)
    { GemmArgs g = ga_default(); g.A = y; g.Bm = WkvT[l]; g.Cf = kv;
      g.M = M; g.Nn = 2 * HD_; g.K = D_; g.lda = D_; g.ldb = D_; g.ldc = 2 * HD_;
      launch_gemm(stream, g, 1); }
    f32_to_bf16_kernel<<<4096, 256, 0, stream>>>(kv, kvb, (long long)M * 2 * HD_);

    // sT[b,h,r,n] = sum_d WpT[r,d] * k[b,h,n,d]   (NT, batch=B*H)
    { GemmArgs g = ga_default(); g.A = WpT[l]; g.Bm = kvb; g.Cf = sT;
      g.M = R_; g.Nn = N_; g.K = DH_; g.lda = DH_; g.ldb = 2 * HD_; g.ldc = N_;
      g.bn0 = H_; g.bn1 = B_;
      g.sB0 = DH_; g.sB1 = (long long)N_ * 2 * HD_;
      g.sC0 = (long long)R_ * N_; g.sC1 = (long long)H_ * R_ * N_;
      launch_gemm(stream, g, B_ * H_); }
    // p = softmax over n  -> pT bf16 (B,H,R,N)
    softmax_rows_kernel<<<B_ * H_ * R_, 256, 0, stream>>>(sT, pT, N_);

    // gk_raw[r,d] = sum_n pT[r,n] * k[n,d]   (NN, K=4096)
    { GemmArgs g = ga_default(); g.A = pT; g.Bm = kvb; g.Cf = gkraw;
      g.M = R_; g.Nn = DH_; g.K = N_; g.lda = N_; g.ldb = 2 * HD_; g.ldc = DH_;
      g.flags = F_BNN; g.bn0 = H_; g.bn1 = B_;
      g.sA0 = (long long)R_ * N_; g.sA1 = (long long)H_ * R_ * N_;
      g.sB0 = DH_; g.sB1 = (long long)N_ * 2 * HD_;
      g.sC0 = (long long)R_ * DH_; g.sC1 = (long long)H_ * R_ * DH_;
      launch_gemm(stream, g, B_ * H_); }
    { GemmArgs g = ga_default(); g.A = pT; g.Bm = kvb + HD_; g.Cf = gvraw;
      g.M = R_; g.Nn = DH_; g.K = N_; g.lda = N_; g.ldb = 2 * HD_; g.ldc = DH_;
      g.flags = F_BNN; g.bn0 = H_; g.bn1 = B_;
      g.sA0 = (long long)R_ * N_; g.sA1 = (long long)H_ * R_ * N_;
      g.sB0 = DH_; g.sB1 = (long long)N_ * 2 * HD_;
      g.sC0 = (long long)R_ * DH_; g.sC1 = (long long)H_ * R_ * DH_;
      launch_gemm(stream, g, B_ * H_); }
    // gk/gv = gln(raw) -> bf16
    layernorm_kernel<<<B_ * H_ * R_, 64, 0, stream>>>(gkraw, gkb, gln_g + l * DH_,
                                                      gln_b + l * DH_, DH_, 1, 0, DH_, DH_);
    layernorm_kernel<<<B_ * H_ * R_, 64, 0, stream>>>(gvraw, gvb, gln_g + l * DH_,
                                                      gln_b + l * DH_, DH_, 1, 0, DH_, DH_);
    // lk/lv = lln(k/v): rows are (b,n,h); skip the v half every H rows
    layernorm_kernel<<<B_ * N_ * H_, 64, 0, stream>>>(kv, lk, lln_g + l * DH_,
                                                      lln_b + l * DH_, DH_, H_, HD_, DH_, DH_);
    layernorm_kernel<<<B_ * N_ * H_, 64, 0, stream>>>(kv + HD_, lv, lln_g + l * DH_,
                                                      lln_b + l * DH_, DH_, H_, HD_, DH_, DH_);

    // sim_l[b,h,w,i,j] = lq_i . lk_j   (NT, batch = B*H*NW)
    { GemmArgs g = ga_default(); g.A = qb; g.Bm = lk; g.Cf = sim;
      g.M = WS_; g.Nn = WS_; g.K = DH_; g.lda = HD_; g.ldb = HD_; g.ldc = WS_ + R_;
      g.bn0 = NW_; g.bn1 = H_;
      g.sA0 = (long long)WS_ * HD_; g.sA1 = DH_; g.sA2 = (long long)N_ * HD_;
      g.sB0 = (long long)WS_ * HD_; g.sB1 = DH_; g.sB2 = (long long)N_ * HD_;
      g.sC0 = (long long)WS_ * (WS_ + R_); g.sC1 = (long long)NW_ * WS_ * (WS_ + R_);
      g.sC2 = (long long)H_ * NW_ * WS_ * (WS_ + R_);
      launch_gemm(stream, g, B_ * H_ * NW_); }
    // sim_g[b,h,w,i,r] = lq_i . gk_r
    { GemmArgs g = ga_default(); g.A = qb; g.Bm = gkb; g.Cf = sim + WS_;
      g.M = WS_; g.Nn = R_; g.K = DH_; g.lda = HD_; g.ldb = DH_; g.ldc = WS_ + R_;
      g.bn0 = NW_; g.bn1 = H_;
      g.sA0 = (long long)WS_ * HD_; g.sA1 = DH_; g.sA2 = (long long)N_ * HD_;
      g.sB0 = 0; g.sB1 = (long long)R_ * DH_; g.sB2 = (long long)H_ * R_ * DH_;
      g.sC0 = (long long)WS_ * (WS_ + R_); g.sC1 = (long long)NW_ * WS_ * (WS_ + R_);
      g.sC2 = (long long)H_ * NW_ * WS_ * (WS_ + R_);
      launch_gemm(stream, g, B_ * H_ * NW_); }
    // attn = softmax(sim) over 384 -> bf16
    softmax_rows_kernel<<<B_ * H_ * N_, 256, 0, stream>>>(sim, attn, WS_ + R_);

    // o = a_l @ lv   (NN), then o += a_g @ gv  -> bf16 (B,N,H,DH)
    { GemmArgs g = ga_default(); g.A = attn; g.Bm = lv; g.Cf = o;
      g.M = WS_; g.Nn = DH_; g.K = WS_; g.lda = WS_ + R_; g.ldb = HD_; g.ldc = HD_;
      g.flags = F_BNN; g.bn0 = NW_; g.bn1 = H_;
      g.sA0 = (long long)WS_ * (WS_ + R_); g.sA1 = (long long)NW_ * WS_ * (WS_ + R_);
      g.sA2 = (long long)H_ * NW_ * WS_ * (WS_ + R_);
      g.sB0 = (long long)WS_ * HD_; g.sB1 = DH_; g.sB2 = (long long)N_ * HD_;
      g.sC0 = (long long)WS_ * HD_; g.sC1 = DH_; g.sC2 = (long long)N_ * HD_;
      launch_gemm(stream, g, B_ * H_ * NW_); }
    { GemmArgs g = ga_default(); g.A = attn + WS_; g.Bm = gvb; g.Cb = ob; g.resid = o;
      g.M = WS_; g.Nn = DH_; g.K = R_; g.lda = WS_ + R_; g.ldb = DH_; g.ldc = HD_;
      g.flags = F_BNN | F_RES | F_BF16OUT; g.bn0 = NW_; g.bn1 = H_;
      g.sA0 = (long long)WS_ * (WS_ + R_); g.sA1 = (long long)NW_ * WS_ * (WS_ + R_);
      g.sA2 = (long long)H_ * NW_ * WS_ * (WS_ + R_);
      g.sB0 = 0; g.sB1 = (long long)R_ * DH_; g.sB2 = (long long)H_ * R_ * DH_;
      g.sC0 = (long long)WS_ * HD_; g.sC1 = DH_; g.sC2 = (long long)N_ * HD_;
      launch_gemm(stream, g, B_ * H_ * NW_); }

    // h = h + ob @ Wo + bo
    { GemmArgs g = ga_default(); g.A = ob; g.Bm = WoT[l]; g.Cf = h; g.resid = h;
      g.bias = bo + l * D_; g.M = M; g.Nn = D_; g.K = HD_;
      g.lda = HD_; g.ldb = HD_; g.ldc = D_; g.flags = F_BIAS | F_RES;
      launch_gemm(stream, g, 1); }

    // FFN: h = h + gelu(ln2(h) @ W1 + b1) @ W2 + b2
    layernorm_kernel<<<M, 256, 0, stream>>>(h, y, ln2_g + l * D_, ln2_b + l * D_,
                                            D_, 1, 0, D_, D_);
    { GemmArgs g = ga_default(); g.A = y; g.Bm = W1T[l]; g.Cb = a1;
      g.bias = b1 + l * FF_; g.M = M; g.Nn = FF_; g.K = D_;
      g.lda = D_; g.ldb = D_; g.ldc = FF_; g.flags = F_BIAS | F_GELU | F_BF16OUT;
      launch_gemm(stream, g, 1); }
    { GemmArgs g = ga_default(); g.A = a1; g.Bm = W2T[l]; g.Cf = h; g.resid = h;
      g.bias = b2 + l * D_; g.M = M; g.Nn = D_; g.K = FF_;
      g.lda = FF_; g.ldb = FF_; g.ldc = D_; g.flags = F_BIAS | F_RES;
      launch_gemm(stream, g, 1); }
  }

  // logits = lnf(h) @ Wlog + blog
  layernorm_kernel<<<M, 256, 0, stream>>>(h, y, lnf_g, lnf_b, D_, 1, 0, D_, D_);
  { GemmArgs g = ga_default(); g.A = y; g.Bm = WlogT; g.Cf = logits; g.bias = blog;
    g.M = M; g.Nn = V_; g.K = D_; g.lda = D_; g.ldb = D_; g.ldc = V_;
    g.flags = F_BIAS; launch_gemm(stream, g, 1); }

  // out = logits.reshape(B, N*V) @ Wfin + bfin   (deterministic 2-stage reduction)
  final_partial_kernel<<<FCHUNKS, 256, 0, stream>>>(logits, Wfin, part);
  final_reduce_kernel<<<1, 64, 0, stream>>>(part, bfin, out);
}
